// GATLayer_36988258353779
// MI455X (gfx1250) — compile-verified, matrix operand order checked by reading, not run on hardware
//
#include <hip/hip_runtime.h>
#include <math.h>

typedef __attribute__((ext_vector_type(16))) _Float16 v16h;
typedef __attribute__((ext_vector_type(8)))  _Float16 v8h;
typedef __attribute__((ext_vector_type(8)))  float    v8f;

static constexpr float NEG_SLOPE = 0.2f;

// ---------------------------------------------------------------------------
// utility: fill
// ---------------------------------------------------------------------------
__global__ void fill_kernel(float* __restrict__ p, long long n, float v) {
  long long i = blockIdx.x * (long long)blockDim.x + threadIdx.x;
  if (i < n) p[i] = v;
}

// f32 -> f16 elementwise
__global__ void cvt_f16_kernel(const float* __restrict__ src,
                               _Float16* __restrict__ dst, long long n) {
  long long i = blockIdx.x * (long long)blockDim.x + threadIdx.x;
  if (i < n) dst[i] = (_Float16)src[i];
}

// W[K,N] f32 -> Wt[N,K] f16 (transpose + convert; W is tiny, L2-resident)
__global__ void transpose_cvt_kernel(const float* __restrict__ W,
                                     _Float16* __restrict__ Wt, int K, int N) {
  int t = blockIdx.x * blockDim.x + threadIdx.x;
  if (t >= K * N) return;
  int k = t / N, nn = t % N;
  Wt[(size_t)nn * K + k] = (_Float16)W[t];
}

// ---------------------------------------------------------------------------
// float atomic max via monotone int/uint punning (exact IEEE max)
// ---------------------------------------------------------------------------
__device__ __forceinline__ void atomicMaxF(float* addr, float val) {
  if (val >= 0.0f) {
    atomicMax((int*)addr, __float_as_int(val));
  } else {
    atomicMin((unsigned int*)addr, __float_as_uint(val));
  }
}

// ---------------------------------------------------------------------------
// WMMA GEMM:  O[M,N] = A[M,K] @ B[K,N], A f16 row-major, Bt = B^T f16 [N,K].
// One wave per 16x16 tile; wave w owns n-tile w; K fully unrolled (4 WMMAs).
// A 16-bit operand layout (ISA 7.12.2): lane L holds row L%16; K interleaved:
//   halves 0-7:  K = ko..ko+7          (ko = L<16 ? 0 : 8)   -> one b128 load
//   halves 8-15: K = 16+ko..16+ko+7                          -> one b128 load
// B symmetric: lane L holds column L%16, same K mapping (from Bt rows).
// D f32 layout: VGPR r -> row (L<16 ? r : 8+r), column L%16.
// ---------------------------------------------------------------------------
template <int K, int N>
__global__ void wmma_gemm_f16_kernel(const _Float16* __restrict__ A,
                                     const _Float16* __restrict__ Bt,
                                     float* __restrict__ O, int M) {
  const int lane  = threadIdx.x & 31;
  const int wave  = threadIdx.x >> 5;
  const int mTile = blockIdx.x;
  const int nTile = wave;                 // blockDim.x == (N/16)*32

  const int lm   = lane & 15;
  const int ko   = (lane < 16) ? 0 : 8;
  const int ncol = nTile * 16 + lm;
  int row        = mTile * 16 + lm;
  if (row >= M) row = M - 1;              // safe dup (never stored OOB)

  const _Float16* arow = A  + (size_t)row  * K;
  const _Float16* brow = Bt + (size_t)ncol * K;
  v8f acc = {};

#pragma unroll
  for (int kk = 0; kk < K; kk += 32) {
    v8h alo = *(const v8h*)(arow + kk + ko);
    v8h ahi = *(const v8h*)(arow + kk + 16 + ko);
    v8h blo = *(const v8h*)(brow + kk + ko);
    v8h bhi = *(const v8h*)(brow + kk + 16 + ko);
    v16h a = __builtin_shufflevector(alo, ahi, 0, 1, 2, 3, 4, 5, 6, 7,
                                     8, 9, 10, 11, 12, 13, 14, 15);
    v16h b = __builtin_shufflevector(blo, bhi, 0, 1, 2, 3, 4, 5, 6, 7,
                                     8, 9, 10, 11, 12, 13, 14, 15);
    acc = __builtin_amdgcn_wmma_f32_16x16x32_f16(
        /*neg_a=*/false, a, /*neg_b=*/false, b,
        /*c_mod=*/(short)0, acc, /*reuse_a=*/false, /*reuse_b=*/false);
  }

  const int mBase = mTile * 16 + ((lane < 16) ? 0 : 8);
#pragma unroll
  for (int r = 0; r < 8; ++r) {
    int rr = mBase + r;
    if (rr < M) O[(size_t)rr * N + ncol] = acc[r];
  }
}

// ---------------------------------------------------------------------------
// per-node attention scores: al_s[n,h] = <xp[n,h,:], a_src[h,:]>, same for dst
// ---------------------------------------------------------------------------
__global__ void node_scores_kernel(const float* __restrict__ xp,
                                   const float* __restrict__ a_src,
                                   const float* __restrict__ a_dst,
                                   float* __restrict__ als,
                                   float* __restrict__ ald,
                                   int n, int H, int C) {
  int t = blockIdx.x * blockDim.x + threadIdx.x;
  if (t >= n * H) return;
  int node = t / H, h = t % H;
  const float* rowp = xp + (size_t)node * H * C + (size_t)h * C;
  float ss = 0.0f, sd = 0.0f;
  for (int c = 0; c < C; ++c) {
    float xv = rowp[c];
    ss += xv * a_src[h * C + c];
    sd += xv * a_dst[h * C + c];
  }
  als[t] = ss;
  ald[t] = sd;
}

// ---------------------------------------------------------------------------
// edge pass 1: leaky-relu logits + segment max over dst
// ---------------------------------------------------------------------------
__global__ void edge_logits_max_kernel(const int* __restrict__ ei, int E, int Etot,
                                       const float* __restrict__ als,
                                       const float* __restrict__ ald,
                                       float* __restrict__ m,
                                       float* __restrict__ logits, int H) {
  int e = blockIdx.x * blockDim.x + threadIdx.x;
  if (e >= Etot) return;
  int s = (e < E) ? ei[e]     : (e - E);
  int d = (e < E) ? ei[E + e] : (e - E);
  for (int h = 0; h < H; ++h) {
    float l = als[s * H + h] + ald[d * H + h];
    l = (l > 0.0f) ? l : NEG_SLOPE * l;
    logits[(size_t)e * H + h] = l;
    atomicMaxF(&m[d * H + h], l);
  }
}

// ---------------------------------------------------------------------------
// edge pass 2: e = exp(logit - m[dst]); segment sum over dst (in place)
// ---------------------------------------------------------------------------
__global__ void edge_exp_sum_kernel(const int* __restrict__ ei, int E, int Etot,
                                    const float* __restrict__ m,
                                    float* __restrict__ logits,
                                    float* __restrict__ ssum, int H) {
  int e = blockIdx.x * blockDim.x + threadIdx.x;
  if (e >= Etot) return;
  int d = (e < E) ? ei[E + e] : (e - E);
  for (int h = 0; h < H; ++h) {
    float ex = __expf(logits[(size_t)e * H + h] - m[d * H + h]);
    logits[(size_t)e * H + h] = ex;
    atomicAdd(&ssum[d * H + h], ex);
  }
}

// ---------------------------------------------------------------------------
// edge pass 3: alpha = e / s[dst]; out[dst,:] += alpha * xp[src,:]
// one thread per (edge, channel); HC consecutive threads share an edge
// ---------------------------------------------------------------------------
__global__ void edge_aggregate_kernel(const int* __restrict__ ei, int E, int Etot,
                                      const float* __restrict__ eexp,
                                      const float* __restrict__ ssum,
                                      const float* __restrict__ xp,
                                      float* __restrict__ out,
                                      float* __restrict__ alphaOut,
                                      int H, int C) {
  const int HC = H * C;
  long long g = blockIdx.x * (long long)blockDim.x + threadIdx.x;
  long long total = (long long)Etot * HC;
  if (g >= total) return;
  int e = (int)(g / HC);
  int c = (int)(g % HC);
  int h = c / C;
  int s = (e < E) ? ei[e]     : (e - E);
  int d = (e < E) ? ei[E + e] : (e - E);
  float alpha = eexp[(size_t)e * H + h] / ssum[d * H + h];
  if (alphaOut != nullptr && (c % C) == 0) alphaOut[(size_t)e * H + h] = alpha;
  atomicAdd(&out[(size_t)d * HC + c], alpha * xp[(size_t)s * HC + c]);
}

// ---------------------------------------------------------------------------
// h16 = (f16) relu(acc + b)      (feeds GEMM2 directly in f16)
// ---------------------------------------------------------------------------
__global__ void relu_bias_f16_kernel(const float* __restrict__ acc,
                                     const float* __restrict__ b,
                                     _Float16* __restrict__ out,
                                     long long n, int D) {
  long long t = blockIdx.x * (long long)blockDim.x + threadIdx.x;
  if (t >= n) return;
  float v = acc[t] + b[(int)(t % D)];
  out[t] = (_Float16)((v > 0.0f) ? v : 0.0f);
}

__global__ void bias_add_kernel(float* __restrict__ x, const float* __restrict__ b,
                                long long n, int D) {
  long long t = blockIdx.x * (long long)blockDim.x + threadIdx.x;
  if (t >= n) return;
  x[t] += b[(int)(t % D)];
}

// ---------------------------------------------------------------------------
// edge_index with self loops, emitted as floats (exact: indices < 2^24)
// ---------------------------------------------------------------------------
__global__ void edge_out_kernel(const int* __restrict__ ei, int E, int Etot,
                                float* __restrict__ o) {
  int t = blockIdx.x * blockDim.x + threadIdx.x;
  if (t >= 2 * Etot) return;
  int row = t / Etot, i = t % Etot;
  int val = (i < E) ? ei[row * E + i] : (i - E);
  o[t] = (float)val;
}

// ---------------------------------------------------------------------------
// launch
// ---------------------------------------------------------------------------
extern "C" void kernel_launch(void* const* d_in, const int* in_sizes, int n_in,
                              void* d_out, int out_size, void* d_ws, size_t ws_size,
                              hipStream_t stream) {
  const float* x    = (const float*)d_in[0];
  const int*   ei   = (const int*)d_in[1];     // edge_index flattened [2,E]
  const float* W1   = (const float*)d_in[2];
  const float* as1  = (const float*)d_in[3];
  const float* ad1  = (const float*)d_in[4];
  const float* b1   = (const float*)d_in[5];
  const float* W2   = (const float*)d_in[6];
  const float* as2  = (const float*)d_in[7];
  const float* ad2  = (const float*)d_in[8];
  const float* b2   = (const float*)d_in[9];

  const int D_IN = 128, H1 = 4, C1 = 32, HC1 = 128, H2 = 1, C2 = 64, HC2 = 64;
  const int n    = in_sizes[0] / D_IN;         // 50000
  const int E    = in_sizes[1] / 2;            // 800000
  const int Etot = E + n;                      // 850000

  // ----- output layout (floats, return-order concat) -----
  float* out     = (float*)d_out;
  float* o_edge  = out;                          // 2*Etot
  float* o_alpha = out + 2 * (size_t)Etot;       // Etot
  float* o_feat  = o_alpha + (size_t)Etot;       // n*64

  // ----- workspace layout (float-granularity offsets) -----
  float* ws    = (float*)d_ws;
  float* xp1   = ws;                                       // n*128 f32
  float* als1  = ws + (size_t)n * HC1;                     // n*4
  float* ald1  = als1 + (size_t)n * H1;
  float* m1    = ald1 + (size_t)n * H1;
  float* s1v   = m1   + (size_t)n * H1;
  float* elog1 = s1v  + (size_t)n * H1;                    // Etot*4
  float* h1    = elog1 + (size_t)Etot * H1;                // n*128 f32 accumulator
  _Float16* habuf = (_Float16*)(h1 + (size_t)n * HC1);     // n*128 f16 (xh, then h1h)
  _Float16* W1t   = habuf + (size_t)n * HC1;               // 128*128 f16
  _Float16* W2t   = W1t + 128 * 128;                       // 64*128 f16
  // layer-2 buffers alias the (dead after layer 1) xp1/scores region
  float* xp2   = ws;                                       // n*64
  float* als2  = ws + (size_t)n * HC2;                     // n
  float* ald2  = als2 + n;
  float* m2    = ald2 + n;
  float* s2v   = m2 + n;
  float* elog2 = s2v + n;                                  // Etot

  auto nb = [](long long cnt, int bs) { return (unsigned)((cnt + bs - 1) / bs); };
  const float NEG_INF = -__builtin_inff();
  const int mTiles = (n + 15) / 16;

  // ===== prep: f16 operands =====
  cvt_f16_kernel<<<nb((long long)n * D_IN, 256), 256, 0, stream>>>(x, habuf, (long long)n * D_IN);
  transpose_cvt_kernel<<<nb(128 * 128, 256), 256, 0, stream>>>(W1, W1t, 128, 128);
  transpose_cvt_kernel<<<nb(128 * 64, 256), 256, 0, stream>>>(W2, W2t, 128, 64);

  // ===== layer 1 =====
  fill_kernel<<<nb((long long)n * H1, 256), 256, 0, stream>>>(m1, (long long)n * H1, NEG_INF);
  fill_kernel<<<nb((long long)n * H1, 256), 256, 0, stream>>>(s1v, (long long)n * H1, 0.0f);
  fill_kernel<<<nb((long long)n * HC1, 256), 256, 0, stream>>>(h1, (long long)n * HC1, 0.0f);

  wmma_gemm_f16_kernel<128, 128><<<mTiles, 256, 0, stream>>>(habuf, W1t, xp1, n);
  node_scores_kernel<<<nb((long long)n * H1, 256), 256, 0, stream>>>(xp1, as1, ad1, als1, ald1, n, H1, C1);
  edge_logits_max_kernel<<<nb(Etot, 256), 256, 0, stream>>>(ei, E, Etot, als1, ald1, m1, elog1, H1);
  edge_exp_sum_kernel<<<nb(Etot, 256), 256, 0, stream>>>(ei, E, Etot, m1, elog1, s1v, H1);
  edge_aggregate_kernel<<<nb((long long)Etot * HC1, 256), 256, 0, stream>>>(
      ei, E, Etot, elog1, s1v, xp1, h1, nullptr, H1, C1);
  // h (f16) = relu(acc + b1), overwriting the x-f16 staging buffer
  relu_bias_f16_kernel<<<nb((long long)n * HC1, 256), 256, 0, stream>>>(
      h1, b1, habuf, (long long)n * HC1, HC1);

  // ===== layer 2 =====
  fill_kernel<<<nb(n, 256), 256, 0, stream>>>(m2, n, NEG_INF);
  fill_kernel<<<nb(n, 256), 256, 0, stream>>>(s2v, n, 0.0f);
  fill_kernel<<<nb((long long)n * HC2, 256), 256, 0, stream>>>(o_feat, (long long)n * HC2, 0.0f);

  wmma_gemm_f16_kernel<128, 64><<<mTiles, 128, 0, stream>>>(habuf, W2t, xp2, n);
  node_scores_kernel<<<nb(n, 256), 256, 0, stream>>>(xp2, as2, ad2, als2, ald2, n, H2, C2);
  edge_logits_max_kernel<<<nb(Etot, 256), 256, 0, stream>>>(ei, E, Etot, als2, ald2, m2, elog2, H2);
  edge_exp_sum_kernel<<<nb(Etot, 256), 256, 0, stream>>>(ei, E, Etot, m2, elog2, s2v, H2);
  edge_aggregate_kernel<<<nb((long long)Etot * HC2, 256), 256, 0, stream>>>(
      ei, E, Etot, elog2, s2v, xp2, o_feat, o_alpha, H2, C2);
  bias_add_kernel<<<nb((long long)n * HC2, 256), 256, 0, stream>>>(o_feat, b2, (long long)n * HC2, HC2);

  // ===== edge_index output =====
  edge_out_kernel<<<nb(2 * (long long)Etot, 256), 256, 0, stream>>>(ei, E, Etot, o_edge);
}
